// MultiheadLSHAttention_919123001907
// MI455X (gfx1250) — matrix operand order repaired and checked by reading, not compiled, not run on hardware
//
#include <hip/hip_runtime.h>

// ---------------------------------------------------------------------------
// CDNA5 (gfx1250) wave32 WMMA implementation of shared-QK multihead attention.
//   q = Q@Wqk^T + b ; k = K@Wqk^T + b ; v = V@Wv^T + bv
//   head h = contiguous flat chunk (pure reshape!), attn per head (4096x64),
//   out = (attn merge) @ Wout^T + bout
// All matmuls use v_wmma_f32_16x16x32_bf16 (fp32 accum), softmax in fp32.
// v3: explicit two-stage pipeline in the GEMM with *named* fragment buffers
//     (no dynamically-indexed arrays -> no scratch/LDS spilling; v2's a[cur]
//     indexing forced the fragments into scratch memory). K hardcoded to 512
//     so trip counts are static and the loop body stays fully register-resident.
// ---------------------------------------------------------------------------

typedef float v8f   __attribute__((ext_vector_type(8)));
typedef __bf16 v16bf __attribute__((ext_vector_type(16)));

union FragBF {
    uint4 u4[2];   // two 16B chunks (8 bf16 each)
    v16bf v;       // 16 bf16 = 8 VGPRs per lane
};

__device__ __forceinline__ unsigned short f2bf(float f) {
    unsigned int u = __float_as_uint(f);
    u += 0x7fffu + ((u >> 16) & 1u);        // round-to-nearest-even
    return (unsigned short)(u >> 16);
}

__device__ __forceinline__ v8f wmma_bf16(const FragBF& a, const FragBF& b, v8f c) {
    return __builtin_amdgcn_wmma_f32_16x16x32_bf16(
        /*neg_a=*/false, a.v, /*neg_b=*/false, b.v,
        /*c_mod=*/(short)0, c, /*reuse_a=*/false, /*reuse_b=*/false);
}

// A fragment (16x32 bf16): per lane 16 contiguous K
__device__ __forceinline__ void load_afrag(FragBF& f, const unsigned short* p) {
    const uint4* q = (const uint4*)p;
    f.u4[0] = q[0];
    f.u4[1] = q[1];
}
// B fragment (32x16 bf16): per lane 16 contiguous K
__device__ __forceinline__ void load_bfrag(FragBF& f, const unsigned short* p) {
    const uint4* q = (const uint4*)p;
    f.u4[0] = q[0];
    f.u4[1] = q[1];
}

// ------------------------------- convert -----------------------------------
__global__ void cvt_f32_bf16(const float* __restrict__ s,
                             unsigned short* __restrict__ d, int n) {
    int i = blockIdx.x * blockDim.x + threadIdx.x;
    int stride = gridDim.x * blockDim.x;
    for (; i < n; i += stride) d[i] = f2bf(s[i]);
}

// ------------------------------- GEMM --------------------------------------
// out(Mx512) = A(Mx512 bf16, row-major) @ W^T (W 512x512 bf16 row-major) + bias
// One wave -> 16x64 tile (4 N-subtiles). Two-stage software pipeline with
// explicit register fragment buffers (a0/b0 and a1/b1), K = 512 fixed.
// mode 0: store bf16 row-major (stride 512)
// mode 1: store bf16 per-head transposed Vt[h][d][t]   (for attention P@V)
// mode 2: store f32 row-major (final output)
__global__ void gemm_wmma(const unsigned short* __restrict__ A,
                          const unsigned short* __restrict__ W,
                          const float* __restrict__ bias,
                          void* __restrict__ out,
                          int mode) {
    constexpr int KD = 512;
    const int lane = threadIdx.x;
    const int half = lane >> 4;    // 0|1
    const int l16  = lane & 15;
    const int row0 = blockIdx.x * 16;
    const int col0 = blockIdx.y * 64;

    v8f acc[4] = {};

    const unsigned short* Abase = A + (size_t)(row0 + l16) * KD + half * 16;
    const unsigned short* Wrow[4];
    #pragma unroll
    for (int t = 0; t < 4; ++t)
        Wrow[t] = W + (size_t)(col0 + 16 * t + l16) * KD + half * 16;

    // stage buffers: distinct named objects, only unroll-constant indexing
    FragBF a0, a1, b0[4], b1[4];
    load_afrag(a0, Abase);
    #pragma unroll
    for (int t = 0; t < 4; ++t) load_bfrag(b0[t], Wrow[t]);

    for (int kk = 0; kk < KD; kk += 64) {
        // issue loads for kk+32 while WMMAs on (a0,b0) run
        load_afrag(a1, Abase + kk + 32);
        #pragma unroll
        for (int t = 0; t < 4; ++t) load_bfrag(b1[t], Wrow[t] + kk + 32);
        #pragma unroll
        for (int t = 0; t < 4; ++t) acc[t] = wmma_bf16(a0, b0[t], acc[t]);

        // issue loads for kk+64 while WMMAs on (a1,b1) run
        if (kk + 64 < KD) {
            load_afrag(a0, Abase + kk + 64);
            #pragma unroll
            for (int t = 0; t < 4; ++t) load_bfrag(b0[t], Wrow[t] + kk + 64);
        }
        #pragma unroll
        for (int t = 0; t < 4; ++t) acc[t] = wmma_bf16(a1, b1[t], acc[t]);
    }

    #pragma unroll
    for (int t = 0; t < 4; ++t) {
        const int e_out = col0 + 16 * t + l16;
        const float bv = bias[e_out];
        #pragma unroll
        for (int r = 0; r < 8; ++r) {
            const int row = row0 + r + 8 * half;
            const float val = acc[t][r] + bv;
            if (mode == 0) {
                ((unsigned short*)out)[(size_t)row * 512 + e_out] = f2bf(val);
            } else if (mode == 1) {
                const int h  = row >> 9;
                const int tt = ((row & 511) << 3) | (e_out >> 6);
                const int dd = e_out & 63;
                ((unsigned short*)out)[(size_t)h * 262144 + (size_t)dd * 4096 + tt] = f2bf(val);
            } else {
                ((float*)out)[(size_t)row * 512 + e_out] = val;
            }
        }
    }
}

// ---------------------------- flash attention ------------------------------
// Per (head, 16-query-row tile): online-softmax over 128 chunks of 32 keys.
// Qp/Kp: bf16 head-major [h][s2][d] (== plain row-major 4096x512)
// Vt   : bf16 transposed [h][d][t]
// O    : bf16 head-major [h][s2][d] (== merged (S,E) layout)
__global__ void attn_wmma(const unsigned short* __restrict__ Qp,
                          const unsigned short* __restrict__ Kp,
                          const unsigned short* __restrict__ Vt,
                          unsigned short* __restrict__ O) {
    __shared__ __align__(16) unsigned short Pld[16 * 32];

    const int lane = threadIdx.x;
    const int half = lane >> 4;
    const int l16  = lane & 15;
    const int h    = blockIdx.y;
    const int row0 = blockIdx.x * 16;

    const unsigned short* qh = Qp + (size_t)h * 262144;
    const unsigned short* kh = Kp + (size_t)h * 262144;
    const unsigned short* vh = Vt + (size_t)h * 262144;
    unsigned short*       oh = O  + (size_t)h * 262144;

    // Q fragments for d=0..31 and d=32..63, register-resident all loop long
    FragBF a0, a1;
    {
        const unsigned short* qrow = qh + (size_t)(row0 + l16) * 64 + half * 16;
        load_afrag(a0, qrow);
        load_afrag(a1, qrow + 32);
    }

    // per-lane V row pointers (fixed d = 16*t + l16, contiguous t)
    const unsigned short* vrow[4];
    #pragma unroll
    for (int t = 0; t < 4; ++t)
        vrow[t] = vh + (size_t)(16 * t + l16) * 4096 + half * 16;

    const unsigned short* krow0 = kh + (size_t)(l16) * 64 + half * 16;       // N tile 0
    const unsigned short* krow1 = kh + (size_t)(16 + l16) * 64 + half * 16;  // N tile 1

    v8f o0 = {}, o1 = {}, o2 = {}, o3 = {};
    float m[8], lsum[8];
    #pragma unroll
    for (int r = 0; r < 8; ++r) { m[r] = -1e30f; lsum[r] = 0.0f; }

    for (int tb = 0; tb < 4096; tb += 32) {
        // ---- batched K-fragment loads, then 4 score WMMAs back-to-back ----
        FragBF kb0, kb1, kb2, kb3;
        load_bfrag(kb0, krow0 + (size_t)tb * 64);        // tile0, d 0..31
        load_bfrag(kb1, krow0 + (size_t)tb * 64 + 32);   // tile0, d 32..63
        load_bfrag(kb2, krow1 + (size_t)tb * 64);        // tile1, d 0..31
        load_bfrag(kb3, krow1 + (size_t)tb * 64 + 32);   // tile1, d 32..63

        v8f s0 = {}, s1 = {};
        s0 = wmma_bf16(a0, kb0, s0);
        s0 = wmma_bf16(a1, kb1, s0);
        s1 = wmma_bf16(a0, kb2, s1);
        s1 = wmma_bf16(a1, kb3, s1);

        // ---- issue V-fragment loads now; they overlap the softmax VALU ----
        FragBF vb0, vb1, vb2, vb3;
        load_bfrag(vb0, vrow[0] + tb);
        load_bfrag(vb1, vrow[1] + tb);
        load_bfrag(vb2, vrow[2] + tb);
        load_bfrag(vb3, vrow[3] + tb);

        // ---- online softmax (rows live across 16-lane halves) ----
        #pragma unroll
        for (int r = 0; r < 8; ++r) {
            float x0 = s0[r] * 0.125f;          // 1/sqrt(64)
            float x1 = s1[r] * 0.125f;
            float mx = fmaxf(x0, x1);
            #pragma unroll
            for (int off = 1; off < 16; off <<= 1)
                mx = fmaxf(mx, __shfl_xor(mx, off));
            const float mn = fmaxf(m[r], mx);
            const float alpha = __expf(m[r] - mn);
            m[r] = mn;
            const float p0 = __expf(x0 - mn);
            const float p1 = __expf(x1 - mn);
            float rs = p0 + p1;
            #pragma unroll
            for (int off = 1; off < 16; off <<= 1)
                rs += __shfl_xor(rs, off);
            lsum[r] = lsum[r] * alpha + rs;
            o0[r] *= alpha; o1[r] *= alpha; o2[r] *= alpha; o3[r] *= alpha;
            const int Mrow = r + 8 * half;
            Pld[Mrow * 32 + l16]      = f2bf(p0);   // C-layout -> LDS (row-major 16x32)
            Pld[Mrow * 32 + 16 + l16] = f2bf(p1);
        }
        __syncthreads();

        // reload P as a 16x32 bf16 A-fragment
        FragBF pa;
        load_afrag(pa, &Pld[l16 * 32 + half * 16]);
        __syncthreads();

        // ---- o += p @ v (V fragments already in registers) ----
        o0 = wmma_bf16(pa, vb0, o0);
        o1 = wmma_bf16(pa, vb1, o1);
        o2 = wmma_bf16(pa, vb2, o2);
        o3 = wmma_bf16(pa, vb3, o3);
    }

    // ---- normalize + store (bf16, head layout == merged (S,E)) ----
    #pragma unroll
    for (int r = 0; r < 8; ++r) {
        const float inv = 1.0f / lsum[r];
        const int row = row0 + r + 8 * half;
        unsigned short* orow = oh + (size_t)row * 64;
        orow[l16]      = f2bf(o0[r] * inv);
        orow[16 + l16] = f2bf(o1[r] * inv);
        orow[32 + l16] = f2bf(o2[r] * inv);
        orow[48 + l16] = f2bf(o3[r] * inv);
    }
}

// ------------------------------- launch ------------------------------------
extern "C" void kernel_launch(void* const* d_in, const int* in_sizes, int n_in,
                              void* d_out, int out_size, void* d_ws, size_t ws_size,
                              hipStream_t stream) {
    (void)in_sizes; (void)n_in; (void)out_size; (void)ws_size;

    const float* Q     = (const float*)d_in[0];
    const float* K     = (const float*)d_in[1];
    const float* V     = (const float*)d_in[2];
    const float* W_qk  = (const float*)d_in[3];
    const float* b_qk  = (const float*)d_in[4];
    const float* W_v   = (const float*)d_in[5];
    const float* b_v   = (const float*)d_in[6];
    const float* W_out = (const float*)d_in[7];
    const float* b_out = (const float*)d_in[8];

    const size_t SE = 4096ull * 512ull;   // 2,097,152
    const size_t EE = 512ull * 512ull;    //   262,144

    unsigned short* Qb    = (unsigned short*)d_ws;
    unsigned short* Kb    = Qb    + SE;
    unsigned short* Vb    = Kb    + SE;
    unsigned short* Wqkb  = Vb    + SE;
    unsigned short* Wvb   = Wqkb  + EE;
    unsigned short* Woutb = Wvb   + EE;
    unsigned short* Qp    = Woutb + EE;   // bf16 projections (head-major == row-major)
    unsigned short* Kp    = Qp    + SE;
    unsigned short* Vt    = Kp    + SE;   // bf16 V, transposed per head [h][d][t]
    unsigned short* Oa    = Vt    + SE;   // bf16 attention output (merged layout)

    // fp32 -> bf16 converts
    cvt_f32_bf16<<<1024, 256, 0, stream>>>(Q, Qb, (int)SE);
    cvt_f32_bf16<<<1024, 256, 0, stream>>>(K, Kb, (int)SE);
    cvt_f32_bf16<<<1024, 256, 0, stream>>>(V, Vb, (int)SE);
    cvt_f32_bf16<<<256,  256, 0, stream>>>(W_qk,  Wqkb,  (int)EE);
    cvt_f32_bf16<<<256,  256, 0, stream>>>(W_v,   Wvb,   (int)EE);
    cvt_f32_bf16<<<256,  256, 0, stream>>>(W_out, Woutb, (int)EE);

    // projections: q, k (shared W_qk), v (stored transposed per head)
    dim3 ggrid(256, 8);   // 4096/16 row tiles x 512/64 col groups
    gemm_wmma<<<ggrid, 32, 0, stream>>>(Qb, Wqkb, b_qk, Qp, 0);
    gemm_wmma<<<ggrid, 32, 0, stream>>>(Kb, Wqkb, b_qk, Kp, 0);
    gemm_wmma<<<ggrid, 32, 0, stream>>>(Vb, Wvb,  b_v,  Vt, 1);

    // flash attention: grid = (256 query tiles, 8 heads)
    attn_wmma<<<dim3(256, 8), 32, 0, stream>>>(Qp, Kp, Vt, Oa);

    // output projection -> fp32 d_out
    gemm_wmma<<<ggrid, 32, 0, stream>>>(Oa, Woutb, b_out, d_out, 2);
}
